// FARGANSub_50062138802934
// MI455X (gfx1250) — matrix operand order repaired and checked by reading, not compiled, not run on hardware
//
#include <hip/hip_runtime.h>
#include <hip/hip_bf16.h>

// ---------------------------------------------------------------------------
// FARGAN subframe step, fused, WMMA f16 (CDNA5 / gfx1250, wave32)
// M=32 batch tile per block: each B (weight) tile feeds TWO wmma ops,
// halving L2 weight traffic vs M=16.
// ---------------------------------------------------------------------------
typedef _Float16 v16h __attribute__((ext_vector_type(16)));
typedef _Float16 h2_t __attribute__((ext_vector_type(2)));
typedef float    v8f  __attribute__((ext_vector_type(8)));
typedef _Float16 h_t;

#define DEVINL __device__ __forceinline__

DEVINL float clamp1f(float x) { return fminf(1.0f, fmaxf(-1.0f, x)); }
DEVINL float sigf(float x)    { return 1.0f / (1.0f + __expf(-x)); }

// ---- tiling ---------------------------------------------------------------
static constexpr int ROWS     = 32;     // batch rows per block (2 M-tiles)
static constexpr int NTHREADS = 512;    // 16 wave32s
static constexpr int NWAVES   = NTHREADS / 32;

// ---- model dims -----------------------------------------------------------
static constexpr int S    = 40;
static constexpr int MEMN = 256;
static constexpr int COND = 80;
static constexpr int TMPD = 164;            // cond(80)+pred(44)+prev(40)
static constexpr int KXC  = 352;            // xcat 328 -> pad32
static constexpr int NF   = 192;            // fwc0 width
static constexpr int KX1  = 288;            // 272 -> pad32
static constexpr int H1   = 160;
static constexpr int NG1  = 480;
static constexpr int KX2  = 256;            // 240 -> pad32
static constexpr int H2   = 128;
static constexpr int NG2  = 384;
static constexpr int KX3  = 224;            // 208 -> pad32
static constexpr int H3   = 128;
static constexpr int NG3  = 384;
static constexpr int KCAT = 704;            // 688 -> pad32
static constexpr int NSK  = 128;
static constexpr int NSIG = 48;             // 40 -> pad16

// ---- workspace (f16 padded weights) offsets, in halfs ---------------------
static constexpr size_t OFS_FWC0    = 0;                         // 192x352
static constexpr size_t OFS_FWC0GLU = OFS_FWC0    + 192*352;     // 192x192
static constexpr size_t OFS_IH1     = OFS_FWC0GLU + 192*192;     // 480x288
static constexpr size_t OFS_HH1     = OFS_IH1     + 480*288;     // 480x160
static constexpr size_t OFS_IH2     = OFS_HH1     + 480*160;     // 384x256
static constexpr size_t OFS_HH2     = OFS_IH2     + 384*256;     // 384x128
static constexpr size_t OFS_IH3     = OFS_HH2     + 384*128;     // 384x224
static constexpr size_t OFS_HH3     = OFS_IH3     + 384*224;     // 384x128
static constexpr size_t OFS_GLU1    = OFS_HH3     + 384*128;     // 160x160
static constexpr size_t OFS_GLU2    = OFS_GLU1    + 160*160;     // 128x128
static constexpr size_t OFS_GLU3    = OFS_GLU2    + 128*128;     // 128x128
static constexpr size_t OFS_SKIPGLU = OFS_GLU3    + 128*128;     // 128x128
static constexpr size_t OFS_SKIP    = OFS_SKIPGLU + 128*128;     // 128x704
static constexpr size_t OFS_SIG     = OFS_SKIP    + 128*704;     // 48x128

// ---- LDS layout (bytes), ROWS=32 ------------------------------------------
static constexpr int O_XCAT   = 0;                               // f16 32x352
static constexpr int O_BUFA   = O_XCAT   + ROWS*KXC*2;           // f32 32x480
static constexpr int O_BUFB   = O_BUFA   + ROWS*480*4;           // f32 32x480
static constexpr int O_ACTT   = O_BUFB   + ROWS*480*4;           // f16 32x192
static constexpr int O_X1     = O_ACTT   + ROWS*192*2;           // f16 32x288
static constexpr int O_HH     = O_X1     + ROWS*KX1*2;           // f16 32x160
static constexpr int O_X2     = O_HH     + ROWS*160*2;           // f16 32x256
static constexpr int O_X3     = O_X2     + ROWS*KX2*2;           // f16 32x224
static constexpr int O_CAT    = O_X3     + ROWS*KX3*2;           // f16 32x704
static constexpr int O_SK     = O_CAT    + ROWS*KCAT*2;          // f16 32x128
static constexpr int O_PREV   = O_SK     + ROWS*128*2;           // f16 32x40
static constexpr int O_FPITCH = O_PREV   + ROWS*S*2;             // f32 32x40
static constexpr int O_GAIN   = O_FPITCH + ROWS*S*4;             // f32 32
static constexpr int O_PG     = O_GAIN   + ROWS*4;               // f32 32x4
static constexpr int SMEM_BYTES = O_PG + ROWS*4*4;               // ~272 KB

// ---------------------------------------------------------------------------
// (ROWS x K f16, LDS, stride Ks) @ W^T,  W is (NP x Ks) f16 row-major in
// global (L2-hot).  Two 16-row M-tiles share each B tile -> 2 wmma per
// pair of global b128 loads.  f32 result to outL (LDS, ROWS x NP).
// A/B use contiguous k order (k = 16*half + s, same in both); D layout per CDNA5 ISA §7.12.2.
// ---------------------------------------------------------------------------
DEVINL void mm32(const h_t* __restrict__ actL, int Ks,
                 const h_t* __restrict__ W, int NP,
                 float* __restrict__ outL) {
  const int lane = threadIdx.x & 31;
  const int wv   = threadIdx.x >> 5;
  const int mA   = lane & 15;             // A: row within tile
  const int kbA  = lane & 16;             // A: +16 K for high half-wave
  const int nB   = lane & 15;             // B: column
  const int kbB  = lane & 16;             // B: +16 K for high half-wave
  const int mD   = (lane & 16) >> 1;      // D: row base (0 or 8)
  const h_t* arow0 = actL + mA * Ks + kbA;          // rows 0..15
  const h_t* arow1 = arow0 + 16 * Ks;               // rows 16..31
  for (int t = wv; t < (NP >> 4); t += NWAVES) {
    const int n0 = t << 4;
    const h_t* wrow = W + (size_t)(n0 + nB) * Ks + kbB;
    v8f acc0 = {};
    v8f acc1 = {};
    for (int k0 = 0; k0 < Ks; k0 += 32) {
      v16h a0, a1, b;
      const h2_t* ap0 = (const h2_t*)(arow0 + k0);
      const h2_t* ap1 = (const h2_t*)(arow1 + k0);
#pragma unroll
      for (int r2 = 0; r2 < 8; ++r2) {
        const int kk2 = r2;                         // K pairs 0,2,...,14
        h2_t p0 = ap0[kk2];
        h2_t p1 = ap1[kk2];
        a0[2 * r2] = p0[0]; a0[2 * r2 + 1] = p0[1];
        a1[2 * r2] = p1[0]; a1[2 * r2 + 1] = p1[1];
      }
      b = *(const v16h*)(wrow + k0);                // 16 contiguous halfs (32B)
      acc0 = __builtin_amdgcn_wmma_f32_16x16x32_f16(
          false, a0, false, b, (short)0, acc0, false, false);
      acc1 = __builtin_amdgcn_wmma_f32_16x16x32_f16(
          false, a1, false, b, (short)0, acc1, false, false);
    }
#pragma unroll
    for (int v = 0; v < 8; ++v) {
      outL[(mD + v) * NP + n0 + nB]        = acc0[v];
      outL[(16 + mD + v) * NP + n0 + nB]   = acc1[v];
    }
  }
}

// ---------------------------------------------------------------------------
struct Params {
  const float* cond; const float* prev_pred; const float* exc_mem;
  const int* period;
  const float* sg1; const float* sg2; const float* sg3; const float* sfwc0;
  const float* W_gain; const float* b_gain;
  const float* W_gout; const float* b_gout;
  const h_t* ws;
  float* o_sig; float* o_exc; float* o_prev;
  float* o_g1; float* o_g2; float* o_g3; float* o_fwc;
};

__global__ void fargan_prep_w(const float* __restrict__ src, h_t* __restrict__ dst,
                              int N, int K, int Kpad, int tot) {
  int i = blockIdx.x * blockDim.x + threadIdx.x;
  if (i >= tot) return;
  int n = i / Kpad, k = i - n * Kpad;
  float v = (k < K && n < N) ? src[n * K + k] : 0.0f;
  dst[i] = (h_t)v;
}

__global__ void __launch_bounds__(NTHREADS)
fargan_step_kernel(Params p) {
  extern __shared__ char smem[];
  h_t*   sXcat = (h_t*)(smem + O_XCAT);
  float* bufA  = (float*)(smem + O_BUFA);
  float* bufB  = (float*)(smem + O_BUFB);
  h_t*   sActT = (h_t*)(smem + O_ACTT);
  h_t*   sX1   = (h_t*)(smem + O_X1);
  h_t*   sHH   = (h_t*)(smem + O_HH);
  h_t*   sX2   = (h_t*)(smem + O_X2);
  h_t*   sX3   = (h_t*)(smem + O_X3);
  h_t*   sCat  = (h_t*)(smem + O_CAT);
  h_t*   sSk   = (h_t*)(smem + O_SK);
  h_t*   sPrev = (h_t*)(smem + O_PREV);
  float* sFp   = (float*)(smem + O_FPITCH);
  float* sGain = (float*)(smem + O_GAIN);
  float* sPg   = (float*)(smem + O_PG);

  const int tid  = threadIdx.x;
  const int base = blockIdx.x * ROWS;

  // ---- gain = exp(clamp(cond) @ W_gain + b_gain) --------------------------
  if (tid < ROWS) {
    const int row = base + tid;
    float s = p.b_gain[0];
    for (int k = 0; k < COND; ++k)
      s += clamp1f(p.cond[row * COND + k]) * p.W_gain[k];
    sGain[tid] = __expf(s);
  }
  __syncthreads();

  // ---- stage inputs: tmp = [cond,pred,prev]; xcat = [state_fwc0, tmp] -----
  for (int i = tid; i < ROWS * COND; i += NTHREADS) {    // cond
    int m = i / COND, k = i - m * COND, row = base + m;
    float c = clamp1f(p.cond[row * COND + k]);
    p.o_fwc[row * TMPD + k] = c;
    sXcat[m * KXC + TMPD + k] = (h_t)c;
  }
  for (int i = tid; i < ROWS * (S + 4); i += NTHREADS) { // pred (gathered)
    int m = i / (S + 4), j = i - m * (S + 4), row = base + m;
    int per = p.period[row];
    int ii = MEMN - per + j - 2;
    if (ii >= MEMN) ii -= per;
    float g = sGain[m];
    float pr = clamp1f(p.exc_mem[row * MEMN + ii] / (1e-5f + g));
    if (j >= 2 && j < S + 2) sFp[m * S + (j - 2)] = pr;  // fpitch
    p.o_fwc[row * TMPD + COND + j] = pr;
    sXcat[m * KXC + TMPD + COND + j] = (h_t)pr;
  }
  for (int i = tid; i < ROWS * S; i += NTHREADS) {       // prev
    int m = i / S, j = i - m * S, row = base + m;
    float g = sGain[m];
    float pv = clamp1f(p.exc_mem[row * MEMN + (MEMN - S) + j] / (1e-5f + g));
    sPrev[m * S + j] = (h_t)pv;
    p.o_fwc[row * TMPD + COND + S + 4 + j] = pv;
    sXcat[m * KXC + TMPD + COND + S + 4 + j] = (h_t)pv;
  }
  for (int i = tid; i < ROWS * TMPD; i += NTHREADS) {    // state_fwc0
    int m = i / TMPD, k = i - m * TMPD, row = base + m;
    sXcat[m * KXC + k] = (h_t)p.sfwc0[row * TMPD + k];
  }
  for (int i = tid; i < ROWS * (KXC - 2 * TMPD); i += NTHREADS) { // K pad -> 0
    int m = i / (KXC - 2 * TMPD), k = i - m * (KXC - 2 * TMPD);
    sXcat[m * KXC + 2 * TMPD + k] = (h_t)0.0f;
  }
  __syncthreads();

  // ---- fwc0: t = tanh(xcat @ W.T); out = clamp(glu(t)) --------------------
  mm32(sXcat, KXC, p.ws + OFS_FWC0, NF, bufA);
  __syncthreads();
  for (int i = tid; i < ROWS * NF; i += NTHREADS) {
    float t = tanhf(bufA[i]);
    bufA[i] = t;
    sActT[i] = (h_t)t;
  }
  __syncthreads();
  mm32(sActT, NF, p.ws + OFS_FWC0GLU, NF, bufB);
  __syncthreads();
  for (int i = tid; i < ROWS * NF; i += NTHREADS) {
    int m = i / NF, n = i - m * NF;
    float o = clamp1f(bufA[i] * sigf(bufB[i]));
    sX1[m * KX1 + n] = (h_t)o;            // gru1 input cols [0,192)
    sCat[m * KCAT + 416 + n] = (h_t)o;    // cat cols [416,608)
  }
  __syncthreads();

  // ---- pg = sigmoid(fwc0_out @ W_gout.T + b) ------------------------------
  if (tid < ROWS * 4) {
    int m = tid >> 2, c = tid & 3;
    float s = p.b_gout[c];
    for (int k = 0; k < NF; ++k)
      s += (float)sCat[m * KCAT + 416 + k] * p.W_gout[c * NF + k];
    sPg[m * 4 + c] = sigf(s);
  }
  __syncthreads();

  // ---- splice pg*fpitch, prev, pads into x1/x2/x3/cat; load h1 ------------
  for (int i = tid; i < ROWS * S; i += NTHREADS) {
    int m = i / S, j = i - m * S;
    float f = sFp[m * S + j];
    h_t pv = sPrev[m * S + j];
    sX1[m * KX1 + 192 + j] = (h_t)(sPg[m * 4 + 0] * f);
    sX1[m * KX1 + 232 + j] = pv;
    sX2[m * KX2 + 160 + j] = (h_t)(sPg[m * 4 + 1] * f);
    sX2[m * KX2 + 200 + j] = pv;
    sX3[m * KX3 + 128 + j] = (h_t)(sPg[m * 4 + 2] * f);
    sX3[m * KX3 + 168 + j] = pv;
    sCat[m * KCAT + 608 + j] = (h_t)(sPg[m * 4 + 3] * f);
    sCat[m * KCAT + 648 + j] = pv;
  }
  for (int i = tid; i < ROWS * 16; i += NTHREADS) {      // K pads
    int m = i >> 4, j = i & 15;
    sX1[m * KX1 + 272 + j] = (h_t)0.0f;
    sX2[m * KX2 + 240 + j] = (h_t)0.0f;
    sX3[m * KX3 + 208 + j] = (h_t)0.0f;
    sCat[m * KCAT + 688 + j] = (h_t)0.0f;
  }
  for (int i = tid; i < ROWS * H1; i += NTHREADS) {
    int m = i / H1, n = i - m * H1, row = base + m;
    sHH[m * H1 + n] = (h_t)p.sg1[row * H1 + n];
  }
  __syncthreads();

  // ---- GRU1 ---------------------------------------------------------------
  mm32(sX1, KX1, p.ws + OFS_IH1, NG1, bufA);
  mm32(sHH, H1,  p.ws + OFS_HH1, NG1, bufB);
  __syncthreads();
  for (int i = tid; i < ROWS * H1; i += NTHREADS) {
    int m = i / H1, n = i - m * H1, row = base + m;
    float r = sigf(bufA[m * NG1 + n] + bufB[m * NG1 + n]);
    float z = sigf(bufA[m * NG1 + H1 + n] + bufB[m * NG1 + H1 + n]);
    float c = tanhf(bufA[m * NG1 + 2 * H1 + n] + r * bufB[m * NG1 + 2 * H1 + n]);
    float hn = (1.0f - z) * c + z * p.sg1[row * H1 + n];
    p.o_g1[row * H1 + n] = hn;
    sActT[m * H1 + n] = (h_t)clamp1f(hn);
  }
  __syncthreads();
  mm32(sActT, H1, p.ws + OFS_GLU1, H1, bufA);
  __syncthreads();
  for (int i = tid; i < ROWS * H1; i += NTHREADS) {
    int m = i / H1, n = i - m * H1;
    float hc = (float)sActT[m * H1 + n];
    float o = clamp1f(hc * sigf(bufA[m * H1 + n]));
    sX2[m * KX2 + n] = (h_t)o;
    sCat[m * KCAT + n] = (h_t)o;                          // cat [0,160)
  }
  for (int i = tid; i < ROWS * H2; i += NTHREADS) {
    int m = i / H2, n = i - m * H2, row = base + m;
    sHH[m * H2 + n] = (h_t)p.sg2[row * H2 + n];
  }
  __syncthreads();

  // ---- GRU2 ---------------------------------------------------------------
  mm32(sX2, KX2, p.ws + OFS_IH2, NG2, bufA);
  mm32(sHH, H2,  p.ws + OFS_HH2, NG2, bufB);
  __syncthreads();
  for (int i = tid; i < ROWS * H2; i += NTHREADS) {
    int m = i / H2, n = i - m * H2, row = base + m;
    float r = sigf(bufA[m * NG2 + n] + bufB[m * NG2 + n]);
    float z = sigf(bufA[m * NG2 + H2 + n] + bufB[m * NG2 + H2 + n]);
    float c = tanhf(bufA[m * NG2 + 2 * H2 + n] + r * bufB[m * NG2 + 2 * H2 + n]);
    float hn = (1.0f - z) * c + z * p.sg2[row * H2 + n];
    p.o_g2[row * H2 + n] = hn;
    sActT[m * H2 + n] = (h_t)clamp1f(hn);
  }
  __syncthreads();
  mm32(sActT, H2, p.ws + OFS_GLU2, H2, bufA);
  __syncthreads();
  for (int i = tid; i < ROWS * H2; i += NTHREADS) {
    int m = i / H2, n = i - m * H2;
    float hc = (float)sActT[m * H2 + n];
    float o = clamp1f(hc * sigf(bufA[m * H2 + n]));
    sX3[m * KX3 + n] = (h_t)o;
    sCat[m * KCAT + 160 + n] = (h_t)o;                    // cat [160,288)
  }
  for (int i = tid; i < ROWS * H3; i += NTHREADS) {
    int m = i / H3, n = i - m * H3, row = base + m;
    sHH[m * H3 + n] = (h_t)p.sg3[row * H3 + n];
  }
  __syncthreads();

  // ---- GRU3 ---------------------------------------------------------------
  mm32(sX3, KX3, p.ws + OFS_IH3, NG3, bufA);
  mm32(sHH, H3,  p.ws + OFS_HH3, NG3, bufB);
  __syncthreads();
  for (int i = tid; i < ROWS * H3; i += NTHREADS) {
    int m = i / H3, n = i - m * H3, row = base + m;
    float r = sigf(bufA[m * NG3 + n] + bufB[m * NG3 + n]);
    float z = sigf(bufA[m * NG3 + H3 + n] + bufB[m * NG3 + H3 + n]);
    float c = tanhf(bufA[m * NG3 + 2 * H3 + n] + r * bufB[m * NG3 + 2 * H3 + n]);
    float hn = (1.0f - z) * c + z * p.sg3[row * H3 + n];
    p.o_g3[row * H3 + n] = hn;
    sActT[m * H3 + n] = (h_t)clamp1f(hn);
  }
  __syncthreads();
  mm32(sActT, H3, p.ws + OFS_GLU3, H3, bufA);
  __syncthreads();
  for (int i = tid; i < ROWS * H3; i += NTHREADS) {
    int m = i / H3, n = i - m * H3;
    float hc = (float)sActT[m * H3 + n];
    float o = clamp1f(hc * sigf(bufA[m * H3 + n]));
    sCat[m * KCAT + 288 + n] = (h_t)o;                    // cat [288,416)
  }
  __syncthreads();

  // ---- skip: ct = clamp(tanh(cat @ W_skip.T)); sk = ct*sig(ct@Wg.T) -------
  mm32(sCat, KCAT, p.ws + OFS_SKIP, NSK, bufA);
  __syncthreads();
  for (int i = tid; i < ROWS * NSK; i += NTHREADS) {
    float ct = clamp1f(tanhf(bufA[i]));
    sActT[i] = (h_t)ct;
  }
  __syncthreads();
  mm32(sActT, NSK, p.ws + OFS_SKIPGLU, NSK, bufB);
  __syncthreads();
  for (int i = tid; i < ROWS * NSK; i += NTHREADS) {
    float sk = (float)sActT[i] * sigf(bufB[i]);
    sSk[i] = (h_t)sk;
  }
  __syncthreads();

  // ---- sig_out = tanh(sk @ W_sig.T) * gain --------------------------------
  mm32(sSk, NSK, p.ws + OFS_SIG, NSIG, bufA);
  __syncthreads();
  for (int i = tid; i < ROWS * S; i += NTHREADS) {
    int m = i / S, n = i - m * S, row = base + m;
    float v = tanhf(bufA[m * NSIG + n]) * sGain[m];
    p.o_sig[row * S + n] = v;
    p.o_exc[row * MEMN + (MEMN - S) + n] = v;             // exc_mem_new tail
  }
  // ---- memory shifts ------------------------------------------------------
  for (int i = tid; i < ROWS * (MEMN - S); i += NTHREADS) {
    int m = i / (MEMN - S), c = i - m * (MEMN - S), row = base + m;
    p.o_exc[row * MEMN + c]  = p.exc_mem[row * MEMN + S + c];
    p.o_prev[row * MEMN + c] = p.prev_pred[row * MEMN + S + c];
  }
  for (int i = tid; i < ROWS * S; i += NTHREADS) {
    int m = i / S, n = i - m * S, row = base + m;
    p.o_prev[row * MEMN + (MEMN - S) + n] = sFp[m * S + n];
  }
}

// ---------------------------------------------------------------------------
extern "C" void kernel_launch(void* const* d_in, const int* in_sizes, int n_in,
                              void* d_out, int out_size, void* d_ws, size_t ws_size,
                              hipStream_t stream) {
  (void)n_in; (void)out_size; (void)ws_size;
  const int B = in_sizes[0] / COND;
  h_t* wsh = (h_t*)d_ws;

  // convert + pad all GEMM weights fp32 -> f16 (row-major (Npad, Kpad))
  struct WDesc { int src; size_t ofs; int N, K, NP, KP; };
  const WDesc tbl[14] = {
    {10, OFS_FWC0,    192, 328, 192, 352},
    {11, OFS_FWC0GLU, 192, 192, 192, 192},
    {12, OFS_IH1,     480, 272, 480, 288},
    {13, OFS_HH1,     480, 160, 480, 160},
    {14, OFS_IH2,     384, 240, 384, 256},
    {15, OFS_HH2,     384, 128, 384, 128},
    {16, OFS_IH3,     384, 208, 384, 224},
    {17, OFS_HH3,     384, 128, 384, 128},
    {18, OFS_GLU1,    160, 160, 160, 160},
    {19, OFS_GLU2,    128, 128, 128, 128},
    {20, OFS_GLU3,    128, 128, 128, 128},
    {21, OFS_SKIPGLU, 128, 128, 128, 128},
    {22, OFS_SKIP,    128, 688, 128, 704},
    {23, OFS_SIG,      40, 128,  48, 128},
  };
  for (int i = 0; i < 14; ++i) {
    int tot = tbl[i].NP * tbl[i].KP;
    fargan_prep_w<<<(tot + 255) / 256, 256, 0, stream>>>(
        (const float*)d_in[tbl[i].src], wsh + tbl[i].ofs,
        tbl[i].N, tbl[i].K, tbl[i].KP, tot);
  }

  float* out = (float*)d_out;
  Params p;
  p.cond      = (const float*)d_in[0];
  p.prev_pred = (const float*)d_in[1];
  p.exc_mem   = (const float*)d_in[2];
  p.period    = (const int*)d_in[3];
  p.sg1       = (const float*)d_in[4];
  p.sg2       = (const float*)d_in[5];
  p.sg3       = (const float*)d_in[6];
  p.sfwc0     = (const float*)d_in[7];
  p.W_gain    = (const float*)d_in[8];
  p.b_gain    = (const float*)d_in[9];
  p.W_gout    = (const float*)d_in[24];
  p.b_gout    = (const float*)d_in[25];
  p.ws        = wsh;
  size_t o = 0;
  p.o_sig  = out + o; o += (size_t)B * S;
  p.o_exc  = out + o; o += (size_t)B * MEMN;
  p.o_prev = out + o; o += (size_t)B * MEMN;
  p.o_g1   = out + o; o += (size_t)B * H1;
  p.o_g2   = out + o; o += (size_t)B * H2;
  p.o_g3   = out + o; o += (size_t)B * H3;
  p.o_fwc  = out + o;

  hipFuncSetAttribute((const void*)fargan_step_kernel,
                      hipFuncAttributeMaxDynamicSharedMemorySize, SMEM_BYTES);
  fargan_step_kernel<<<B / ROWS, NTHREADS, SMEM_BYTES, stream>>>(p);
}
